// EdgeConvBlock_27943057227832
// MI455X (gfx1250) — compile-verified, matrix operand order checked by reading, not run on hardware
//
#include <hip/hip_runtime.h>

// ---------------------------------------------------------------------------
// EdgeConv block for MI455X (gfx1250, wave32, WMMA f16 16x16x32).
//
// Pipeline (all on `stream`):
//   k_prep      : W -> f16 copies in ws, zero BN stat accumulators
//   k_knn       : 16-NN indices per point (LDS-staged coords, u64 key top-k)
//   k_layer0    : gather + edge feats -> WMMA(16x128 @ 128x64) -> y0(f16) + stats
//   k_finalize  : BN affine (scale, shift) from (sum, sumsq)
//   k_mid  (x2) : bn+relu fused into A-frag load -> WMMA(16x64 @ 64x64) in-place
//   k_sc        : shortcut features @ scW^T (16 points per wave) + stats
//   k_final     : bn+relu(y2) mean over k + bn+relu(sc) -> out (fp32)
// ---------------------------------------------------------------------------

typedef __attribute__((ext_vector_type(16))) _Float16 v16h;
typedef __attribute__((ext_vector_type(8)))  _Float16 h8;
typedef __attribute__((ext_vector_type(8)))  float    v8f;
typedef __attribute__((ext_vector_type(4)))  float    f4;

#define NCLOUD 128
#define PPTS   512
#define NP     (NCLOUD * PPTS)   // 65536 points
#define KNN    16
#define CIN    64

// ---- helpers ---------------------------------------------------------------

__device__ __forceinline__ v16h combine16(h8 lo, h8 hi) {
  v16h v;
#pragma unroll
  for (int i = 0; i < 8; ++i) { v[i] = lo[i]; v[i + 8] = hi[i]; }
  return v;
}

// 8 consecutive fp32 -> 8 f16
__device__ __forceinline__ h8 pack8(const float* __restrict__ p) {
  f4 a = *(const f4*)p;
  f4 b = *(const f4*)(p + 4);
  h8 r;
#pragma unroll
  for (int i = 0; i < 4; ++i) { r[i] = (_Float16)a[i]; r[i + 4] = (_Float16)b[i]; }
  return r;
}

// 8 consecutive (nbr - center) fp32 -> f16
__device__ __forceinline__ h8 pack8_diff(const float* __restrict__ pn,
                                         const float* __restrict__ pc) {
  f4 a = *(const f4*)pn; f4 b = *(const f4*)(pn + 4);
  f4 c = *(const f4*)pc; f4 d = *(const f4*)(pc + 4);
  h8 r;
#pragma unroll
  for (int i = 0; i < 4; ++i) {
    r[i]     = (_Float16)(a[i] - c[i]);
    r[i + 4] = (_Float16)(b[i] - d[i]);
  }
  return r;
}

// B fragment: lane holds output-channel column o, K chunk starting at kbase.
// W is row-major (o, c) f16 with leading dim ld, so the 8-wide runs at kbase and kbase+16 are each contiguous.
__device__ __forceinline__ v16h load_b(const _Float16* __restrict__ W, int ld,
                                       int o, int kbase) {
  const _Float16* p = W + (size_t)o * ld + kbase;
  return combine16(*(const h8*)p, *(const h8*)(p + 16));
}

__device__ __forceinline__ v8f wmma_f16(v16h a, v16h b, v8f c) {
  return __builtin_amdgcn_wmma_f32_16x16x32_f16(false, a, false, b,
                                                (short)0, c, false, false);
}

// fused bn-affine + relu on an 8-wide f16 chunk (used when loading y as A)
__device__ __forceinline__ h8 bnrelu_chunk(const _Float16* __restrict__ row, int c,
                                           const float* __restrict__ scale,
                                           const float* __restrict__ shift) {
  h8 raw = *(const h8*)(row + c);
  f4 s0 = *(const f4*)(scale + c), s1 = *(const f4*)(scale + c + 4);
  f4 t0 = *(const f4*)(shift + c), t1 = *(const f4*)(shift + c + 4);
  h8 r;
#pragma unroll
  for (int i = 0; i < 4; ++i) {
    r[i]     = (_Float16)fmaxf(fmaf((float)raw[i],     s0[i], t0[i]), 0.f);
    r[i + 4] = (_Float16)fmaxf(fmaf((float)raw[i + 4], s1[i], t1[i]), 0.f);
  }
  return r;
}

// ---- kernel: weight f16 conversion + stat zeroing --------------------------

__global__ void __launch_bounds__(256)
k_prep(const float* __restrict__ W0, const float* __restrict__ W1,
       const float* __restrict__ W2, const float* __restrict__ scW,
       _Float16* __restrict__ W0h, _Float16* __restrict__ W1h,
       _Float16* __restrict__ W2h, _Float16* __restrict__ scWh,
       float* __restrict__ stats) {
  int t = blockIdx.x * 256 + threadIdx.x;
  if (t < 8192) W0h[t] = (_Float16)W0[t];
  if (t < 4096) {
    W1h[t]  = (_Float16)W1[t];
    W2h[t]  = (_Float16)W2[t];
    scWh[t] = (_Float16)scW[t];
  }
  if (t < 512) stats[t] = 0.f;
}

// ---- kernel: 16-NN (matches jax top_k ordering, drops first hit = self) ----

__global__ void __launch_bounds__(128)
k_knn(const float* __restrict__ points, int* __restrict__ idxb) {
  __shared__ float sx[PPTS], sy[PPTS];
  int tid = threadIdx.x;
  int blockPid = blockIdx.x * 4;
  int n = blockPid >> 9;
  const float* pts = points + (size_t)n * PPTS * 2;
  for (int i = tid; i < PPTS; i += 128) { sx[i] = pts[2 * i]; sy[i] = pts[2 * i + 1]; }
  __syncthreads();

  int lane = tid & 31, wv = tid >> 5;
  int pid = blockPid + wv;
  int p = pid & (PPTS - 1);
  float cx = sx[p], cy = sy[p];
  float rp = cx * cx + cy * cy;

  unsigned long long key[16];
#pragma unroll
  for (int j = 0; j < 16; ++j) {
    int q = lane + j * 32;
    float qx = sx[q], qy = sy[q];
    float rq = qx * qx + qy * qy;
    float m  = cx * qx + cy * qy;
    float d  = rp - 2.f * m + rq;           // self -> exactly 0
    unsigned u = __float_as_uint(d);
    u = (u & 0x80000000u) ? ~u : (u | 0x80000000u);   // total order on floats
    key[j] = ((unsigned long long)u << 32) | (unsigned)q;
  }

  for (int it = 0; it < KNN + 1; ++it) {
    unsigned long long lm = ~0ull; int lj = 0;
#pragma unroll
    for (int j = 0; j < 16; ++j)
      if (key[j] < lm) { lm = key[j]; lj = j; }
    unsigned long long gm = lm;
#pragma unroll
    for (int s = 16; s > 0; s >>= 1) {
      unsigned long long o = __shfl_xor(gm, s, 32);
      gm = o < gm ? o : gm;
    }
    if (it > 0 && lane == 0)
      idxb[(size_t)pid * KNN + (it - 1)] = (int)(gm & 0xffffffffu);
    if (lm == gm) key[lj] = ~0ull;          // q unique -> exactly one remover
  }
}

// ---- kernel: layer0 (gather + edge feats + 16x128 @ 128x64 WMMA) -----------

__global__ void __launch_bounds__(128)
k_layer0(const float* __restrict__ feats, const int* __restrict__ idxb,
         const _Float16* __restrict__ W0h, _Float16* __restrict__ ybuf,
         float* __restrict__ stats) {
  __shared__ float lsum[64], lsq[64];
  int tid = threadIdx.x;
  if (tid < 64) { lsum[tid] = 0.f; lsq[tid] = 0.f; }
  __syncthreads();

  int lane = tid & 31, wv = tid >> 5;
  int pid  = blockIdx.x * 4 + wv;
  int n    = pid >> 9;
  int r = lane & 15, h = lane >> 4;

  const float* cen = feats + (size_t)pid * CIN;
  int nb = idxb[(size_t)pid * KNN + r];
  const float* nbr = feats + (((size_t)n << 9) + nb) * CIN;

  v16h A[4];
#pragma unroll
  for (int kt = 0; kt < 2; ++kt) {          // cols 0..63: center broadcast
    int c0 = kt * 32 + h * 8;
    A[kt] = combine16(pack8(cen + c0), pack8(cen + c0 + 16));
  }
#pragma unroll
  for (int kt = 2; kt < 4; ++kt) {          // cols 64..127: nbr - center
    int c0 = (kt - 2) * 32 + h * 8;
    A[kt] = combine16(pack8_diff(nbr + c0, cen + c0),
                      pack8_diff(nbr + c0 + 16, cen + c0 + 16));
  }

  v8f acc[4] = {};
#pragma unroll
  for (int nt = 0; nt < 4; ++nt)
#pragma unroll
    for (int kt = 0; kt < 4; ++kt)
      acc[nt] = wmma_f16(A[kt], load_b(W0h, 128, nt * 16 + r, kt * 32 + h * 8), acc[nt]);

  _Float16* yt = ybuf + (size_t)pid * (KNN * 64);
#pragma unroll
  for (int nt = 0; nt < 4; ++nt) {
    float s = 0.f, q = 0.f;
#pragma unroll
    for (int v = 0; v < 8; ++v) {
      float x = acc[nt][v];
      s += x; q += x * x;
      yt[(size_t)(v + 8 * h) * 64 + nt * 16 + r] = (_Float16)x;   // row-major (k,o)
    }
    s += __shfl_xor(s, 16, 32);
    q += __shfl_xor(q, 16, 32);
    if (h == 0) { atomicAdd(&lsum[nt * 16 + r], s); atomicAdd(&lsq[nt * 16 + r], q); }
  }
  __syncthreads();
  if (tid < 64) { atomicAdd(&stats[tid], lsum[tid]); atomicAdd(&stats[64 + tid], lsq[tid]); }
}

// ---- kernel: BN affine finalize --------------------------------------------

__global__ void k_finalize(const float* __restrict__ stats,
                           const float* __restrict__ g, const float* __restrict__ b,
                           float* __restrict__ affine, float inv_count) {
  int o = threadIdx.x;          // 64 threads
  float mean = stats[o] * inv_count;
  float var  = stats[64 + o] * inv_count - mean * mean;
  float sc   = g[o] * rsqrtf(var + 1e-5f);
  affine[o]      = sc;
  affine[64 + o] = b[o] - mean * sc;
}

// ---- kernel: middle layers (bn+relu fused load, in-place 16x64 @ 64x64) ----

__global__ void __launch_bounds__(128)
k_mid(_Float16* __restrict__ ybuf, const _Float16* __restrict__ Wh,
      const float* __restrict__ affine, float* __restrict__ stats) {
  __shared__ float lsum[64], lsq[64];
  int tid = threadIdx.x;
  if (tid < 64) { lsum[tid] = 0.f; lsq[tid] = 0.f; }
  __syncthreads();

  int lane = tid & 31, wv = tid >> 5;
  int pid  = blockIdx.x * 4 + wv;
  int r = lane & 15, h = lane >> 4;

  _Float16* yt = ybuf + (size_t)pid * (KNN * 64);
  const _Float16* row = yt + (size_t)r * 64;
  const float* scale = affine;
  const float* shift = affine + 64;

  v16h A[2];
#pragma unroll
  for (int kt = 0; kt < 2; ++kt) {
    int c0 = kt * 32 + h * 8;
    A[kt] = combine16(bnrelu_chunk(row, c0, scale, shift),
                      bnrelu_chunk(row, c0 + 16, scale, shift));
  }

  v8f acc[4] = {};
#pragma unroll
  for (int nt = 0; nt < 4; ++nt)
#pragma unroll
    for (int kt = 0; kt < 2; ++kt)
      acc[nt] = wmma_f16(A[kt], load_b(Wh, 64, nt * 16 + r, kt * 32 + h * 8), acc[nt]);

#pragma unroll
  for (int nt = 0; nt < 4; ++nt) {
    float s = 0.f, q = 0.f;
#pragma unroll
    for (int v = 0; v < 8; ++v) {
      float x = acc[nt][v];
      s += x; q += x * x;
      yt[(size_t)(v + 8 * h) * 64 + nt * 16 + r] = (_Float16)x;   // in-place
    }
    s += __shfl_xor(s, 16, 32);
    q += __shfl_xor(q, 16, 32);
    if (h == 0) { atomicAdd(&lsum[nt * 16 + r], s); atomicAdd(&lsq[nt * 16 + r], q); }
  }
  __syncthreads();
  if (tid < 64) { atomicAdd(&stats[tid], lsum[tid]); atomicAdd(&stats[64 + tid], lsq[tid]); }
}

// ---- kernel: shortcut matmul (16 points per wave) --------------------------

__global__ void __launch_bounds__(128)
k_sc(const float* __restrict__ feats, const _Float16* __restrict__ scWh,
     _Float16* __restrict__ scbuf, float* __restrict__ stats) {
  __shared__ float lsum[64], lsq[64];
  int tid = threadIdx.x;
  if (tid < 64) { lsum[tid] = 0.f; lsq[tid] = 0.f; }
  __syncthreads();

  int lane = tid & 31, wv = tid >> 5;
  int base = (blockIdx.x * 4 + wv) * 16;    // base point index (16 per wave)
  int r = lane & 15, h = lane >> 4;

  const float* row = feats + (size_t)(base + r) * CIN;
  v16h A[2];
#pragma unroll
  for (int kt = 0; kt < 2; ++kt) {
    int c0 = kt * 32 + h * 8;
    A[kt] = combine16(pack8(row + c0), pack8(row + c0 + 16));
  }

  v8f acc[4] = {};
#pragma unroll
  for (int nt = 0; nt < 4; ++nt)
#pragma unroll
    for (int kt = 0; kt < 2; ++kt)
      acc[nt] = wmma_f16(A[kt], load_b(scWh, 64, nt * 16 + r, kt * 32 + h * 8), acc[nt]);

#pragma unroll
  for (int nt = 0; nt < 4; ++nt) {
    float s = 0.f, q = 0.f;
#pragma unroll
    for (int v = 0; v < 8; ++v) {
      float x = acc[nt][v];
      s += x; q += x * x;
      scbuf[(size_t)(base + v + 8 * h) * 64 + nt * 16 + r] = (_Float16)x;
    }
    s += __shfl_xor(s, 16, 32);
    q += __shfl_xor(q, 16, 32);
    if (h == 0) { atomicAdd(&lsum[nt * 16 + r], s); atomicAdd(&lsq[nt * 16 + r], q); }
  }
  __syncthreads();
  if (tid < 64) { atomicAdd(&stats[tid], lsum[tid]); atomicAdd(&stats[64 + tid], lsq[tid]); }
}

// ---- kernel: final bn+relu, mean over k, add shortcut ----------------------

__global__ void __launch_bounds__(256)
k_final(const _Float16* __restrict__ ybuf, const _Float16* __restrict__ scbuf,
        const float* __restrict__ aff2, const float* __restrict__ affsc,
        float* __restrict__ out) {
  int gid = blockIdx.x * 256 + threadIdx.x;   // NP*64 outputs
  int pid = gid >> 6, o = gid & 63;
  const _Float16* yt = ybuf + (size_t)pid * (KNN * 64) + o;
  float s2 = aff2[o], t2 = aff2[64 + o];
  float s3 = affsc[o], t3 = affsc[64 + o];
  float m = 0.f;
#pragma unroll
  for (int k = 0; k < KNN; ++k)
    m += fmaxf(fmaf((float)yt[(size_t)k * 64], s2, t2), 0.f);
  m *= (1.f / 16.f);
  float sc = fmaxf(fmaf((float)scbuf[(size_t)pid * 64 + o], s3, t3), 0.f);
  out[gid] = m + sc;
}

// ---- launch ----------------------------------------------------------------

extern "C" void kernel_launch(void* const* d_in, const int* in_sizes, int n_in,
                              void* d_out, int out_size, void* d_ws, size_t ws_size,
                              hipStream_t stream) {
  const float* points = (const float*)d_in[0];
  const float* feats  = (const float*)d_in[1];
  const float* W0  = (const float*)d_in[2];
  const float* g0  = (const float*)d_in[3];
  const float* b0  = (const float*)d_in[4];
  const float* W1  = (const float*)d_in[5];
  const float* g1  = (const float*)d_in[6];
  const float* b1  = (const float*)d_in[7];
  const float* W2  = (const float*)d_in[8];
  const float* g2  = (const float*)d_in[9];
  const float* b2  = (const float*)d_in[10];
  const float* scW = (const float*)d_in[11];
  const float* scg = (const float*)d_in[12];
  const float* scb = (const float*)d_in[13];
  float* out = (float*)d_out;

  char* ws = (char*)d_ws;
  _Float16* W0h   = (_Float16*)(ws);                       // 16 KB
  _Float16* W1h   = (_Float16*)(ws + 16384);               //  8 KB
  _Float16* W2h   = (_Float16*)(ws + 24576);               //  8 KB
  _Float16* scWh  = (_Float16*)(ws + 32768);               //  8 KB
  float*    stats = (float*)(ws + 40960);                  // 4 layers x 128 f32
  float*    aff   = (float*)(ws + 43008);                  // 4 layers x 128 f32
  int*      idxb  = (int*)(ws + 65536);                    //  4 MB
  _Float16* scbuf = (_Float16*)(ws + 65536 + (4u << 20));  //  8 MB
  _Float16* ybuf  = (_Float16*)(ws + 65536 + (12u << 20)); // 128 MB

  const float invNPK = 1.f / (float)(NP * KNN);
  const float invNP  = 1.f / (float)NP;

  k_prep<<<32, 256, 0, stream>>>(W0, W1, W2, scW, W0h, W1h, W2h, scWh, stats);
  k_knn<<<NP / 4, 128, 0, stream>>>(points, idxb);

  k_layer0<<<NP / 4, 128, 0, stream>>>(feats, idxb, W0h, ybuf, stats + 0);
  k_finalize<<<1, 64, 0, stream>>>(stats + 0, g0, b0, aff + 0, invNPK);

  k_mid<<<NP / 4, 128, 0, stream>>>(ybuf, W1h, aff + 0, stats + 128);
  k_finalize<<<1, 64, 0, stream>>>(stats + 128, g1, b1, aff + 128, invNPK);

  k_mid<<<NP / 4, 128, 0, stream>>>(ybuf, W2h, aff + 128, stats + 256);
  k_finalize<<<1, 64, 0, stream>>>(stats + 256, g2, b2, aff + 256, invNPK);

  k_sc<<<NP / 64, 128, 0, stream>>>(feats, scWh, scbuf, stats + 384);
  k_finalize<<<1, 64, 0, stream>>>(stats + 384, scg, scb, aff + 384, invNP);

  k_final<<<(NP * 64) / 256, 256, 0, stream>>>(ybuf, scbuf, aff + 256, aff + 384, out);
}
